// MoleculeEmbedding_82532091560206
// MI455X (gfx1250) — matrix operand order repaired and checked by reading, not compiled, not run on hardware
//
#include <hip/hip_runtime.h>

#define H 768
#define HEADS 8
#define T 3
#define BATCH 128
#define NATOM 48
#define EINTRA 96
#define NPG 49            // nodes per graph
#define EPG 144           // edges per graph
#define NTOT (BATCH * NPG)   // 6272
#define ETOT (BATCH * EPG)   // 18432
#define MOUT (HEADS * H)     // 6144
#define G3H (3 * H)          // 2304
#define KT (H / 32)          // 24 k-tiles

typedef __attribute__((ext_vector_type(16))) __bf16 v16bf;
typedef __attribute__((ext_vector_type(8)))  float  v8f;

// ---------------------------------------------------------------- helpers
__device__ __forceinline__ unsigned short f32_to_bf16_rne(float f) {
  unsigned u = __float_as_uint(f);
  u += 0x7fffu + ((u >> 16) & 1u);
  return (unsigned short)(u >> 16);
}
__device__ __forceinline__ unsigned f2ou(float f) {   // float -> order-preserving uint
  unsigned u = __float_as_uint(f);
  return (u & 0x80000000u) ? ~u : (u | 0x80000000u);
}
__device__ __forceinline__ float ou2f(unsigned u) {
  return (u & 0x80000000u) ? __uint_as_float(u & 0x7fffffffu) : __uint_as_float(~u);
}
__device__ __forceinline__ float sigmoidf(float x) { return 1.0f / (1.0f + __expf(-x)); }

// ---------------------------------------------------------------- f32 -> bf16 (row-major, for A)
__global__ void cvt_bf16_kernel(const float* __restrict__ in,
                                unsigned short* __restrict__ out, long n) {
  long i = (long)blockIdx.x * blockDim.x + threadIdx.x;
  if (i < n) out[i] = f32_to_bf16_rne(in[i]);
}

// ---------------------------------------------------------------- f32 -> bf16 + pack to WMMA B-fragment tiles
// Input  W: (T, H, MOUT) row-major.
// Output per t: [colTile(384)][kTile(24)][lane(32)][j(16)] bf16, where
//   lane<16: col16=lane,    K = kTile*32 + j       (K 0..15 of tile)
//   lane>=16: col16=lane-16, K = kTile*32 + 16 + j (K 16..31 of tile)
__global__ void pack_B_kernel(const float* __restrict__ W,
                              unsigned short* __restrict__ out, long total) {
  long i = (long)blockIdx.x * blockDim.x + threadIdx.x;
  if (i >= total) return;
  const long per = (long)H * MOUT;
  const int  t   = (int)(i / per);
  const long r   = i % per;
  const int  k   = (int)(r / MOUT);
  const int  col = (int)(r % MOUT);
  const int colTile = col >> 4, col16 = col & 15;
  const int kTile = k >> 5, kIn = k & 31;
  const int lane = col16 + ((kIn & 16) ? 16 : 0);
  const int j    = kIn & 15;
  const long off = (long)t * per +
                   ((((long)colTile * KT + kTile) * 32 + lane) * 16 + j);
  out[off] = f32_to_bf16_rne(W[i]);
}

// ---------------------------------------------------------------- WMMA GEMM
// C[z][6272,6144] = A[6272,768](bf16 row-major) * B[z](pre-packed fragments) + bias[z]
#define LDA 40   // padded LDS row stride (halfwords), keeps 16B alignment (40*2=80 % 16 == 0)

union Frag { v16bf v; uint4 q[2]; };

__global__ __launch_bounds__(256)
void gemm_bf16_kernel(const unsigned short* __restrict__ A,
                      const unsigned short* __restrict__ Bsrc,   // packed
                      const unsigned short* __restrict__ Bdst,   // packed
                      const float* __restrict__ bias_src,
                      const float* __restrict__ bias_dst,
                      float* __restrict__ Csrc,
                      float* __restrict__ Cdst) {
  __shared__ __align__(16) unsigned short As[128 * LDA];   // A tile, row-major

  const unsigned short* Bp = blockIdx.z ? Bdst : Bsrc;
  const float* bias = blockIdx.z ? bias_dst : bias_src;
  float* C = blockIdx.z ? Cdst : Csrc;

  const int rowBase = blockIdx.y * 128;
  const int colBase = blockIdx.x * 128;
  const int tid  = threadIdx.x;
  const int lane = tid & 31;
  const int wave = tid >> 5;
  const int waveM = wave >> 2;          // 0..1 -> 64 rows each
  const int waveN = wave & 3;           // 0..3 -> 32 cols each
  const int lrow = lane & 15;
  const int lhi  = lane >> 4;

  const v8f vzero = {0.f,0.f,0.f,0.f,0.f,0.f,0.f,0.f};
  v8f acc[4][2];
#pragma unroll
  for (int mt = 0; mt < 4; ++mt)
#pragma unroll
    for (int nt = 0; nt < 2; ++nt) acc[mt][nt] = vzero;

  const int aRow = tid >> 1, aCol = (tid & 1) * 16;   // 128x32 cooperative A load
  // per-lane packed-B fragment bases for the two n-tiles of this wave
  const unsigned short* Bfrag[2];
#pragma unroll
  for (int nt = 0; nt < 2; ++nt) {
    const int ct = (colBase >> 4) + waveN * 2 + nt;   // global col-tile index
    Bfrag[nt] = Bp + (((long)ct * KT) * 32 + lane) * 16;
  }

  for (int kt = 0; kt < KT; ++kt) {
    const int kk = kt * 32;
    // ---- stage A tile (row-major) into LDS
    const uint4* gA = (const uint4*)(A + (long)(rowBase + aRow) * H + kk + aCol);
    uint4 a0 = gA[0], a1 = gA[1];
    *(uint4*)&As[aRow * LDA + aCol]     = a0;
    *(uint4*)&As[aRow * LDA + aCol + 8] = a1;

    // ---- B fragments straight from global (pre-packed, coalesced b128 pairs)
    Frag bfr[2];
#pragma unroll
    for (int nt = 0; nt < 2; ++nt) {
      const uint4* gB = (const uint4*)(Bfrag[nt] + (long)kt * 512);  // 32 lanes * 16 bf16
      bfr[nt].q[0] = gB[0];
      bfr[nt].q[1] = gB[1];
    }

    if (kt + 1 < KT) {   // prefetch next k-slab -> global_prefetch_b8
      __builtin_prefetch(A + (long)(rowBase + aRow) * H + kk + 32 + aCol, 0, 1);
      __builtin_prefetch(Bfrag[0] + (long)(kt + 1) * 512, 0, 1);
      __builtin_prefetch(Bfrag[1] + (long)(kt + 1) * 512, 0, 1);
    }
    __syncthreads();

    // ---- A fragments (contiguous K order: lane half h holds K 16h..16h+15, as packed B)
    Frag afr[4];
#pragma unroll
    for (int mt = 0; mt < 4; ++mt) {
      const int r = waveM * 64 + mt * 16 + lrow;
      afr[mt].q[0] = *(const uint4*)&As[r * LDA + lhi * 16];       // K 0-7 / 16-23
      afr[mt].q[1] = *(const uint4*)&As[r * LDA + lhi * 16 + 8];   // K 8-15 / 24-31
    }
#pragma unroll
    for (int mt = 0; mt < 4; ++mt)
#pragma unroll
      for (int nt = 0; nt < 2; ++nt)
        acc[mt][nt] = __builtin_amdgcn_wmma_f32_16x16x32_bf16(
            false, afr[mt].v, false, bfr[nt].v, (short)0, acc[mt][nt], false, false);
    __syncthreads();
  }

  // ---- epilogue: bias + store
#pragma unroll
  for (int mt = 0; mt < 4; ++mt) {
#pragma unroll
    for (int nt = 0; nt < 2; ++nt) {
      const int col = colBase + waveN * 32 + nt * 16 + lrow;
      const float bv = bias[col];
#pragma unroll
      for (int r = 0; r < 8; ++r) {
        const int row = rowBase + waveM * 64 + mt * 16 + lhi * 8 + r;
        C[(long)row * MOUT + col] = acc[mt][nt][r] + bv;
      }
    }
  }
}

// ---------------------------------------------------------------- edge logits
// logits[e,h] = sum_d lrelu(fs[src,h,d]+fd[dst,h,d]) * attn_a[h,d]
__global__ void edge_logits_kernel(const float* __restrict__ fs,
                                   const float* __restrict__ fd,
                                   const float* __restrict__ attn_a_t,
                                   const int* __restrict__ src,
                                   const int* __restrict__ dst,
                                   float* __restrict__ logits) {
  const int wid  = (blockIdx.x * blockDim.x + threadIdx.x) >> 5;
  const int lane = threadIdx.x & 31;
  if (wid >= ETOT * HEADS) return;
  const int e = wid >> 3, hh = wid & 7;
  const int s = src[e], d0 = dst[e];
  const float* ps = fs + ((long)s  * HEADS + hh) * H;
  const float* pd = fd + ((long)d0 * HEADS + hh) * H;
  const float* pa = attn_a_t + hh * H;
  float acc = 0.f;
  for (int k = lane; k < H; k += 32) {
    float v = ps[k] + pd[k];
    v = v > 0.f ? v : 0.2f * v;                 // leaky_relu(0.2)
    acc += v * pa[k];
  }
#pragma unroll
  for (int off = 16; off; off >>= 1) acc += __shfl_down(acc, off, 32);
  if (lane == 0) logits[e * HEADS + hh] = acc;
}

// ---------------------------------------------------------------- edge softmax per dst node
// one block per molecule; also emits head-mean attention on virtual edges
__global__ void softmax_kernel(const float* __restrict__ logits,
                               const int* __restrict__ dst,
                               float* __restrict__ att,
                               float* __restrict__ attn_out_t) {
  __shared__ unsigned mx[NPG * HEADS];
  __shared__ float    sm[NPG * HEADS];
  __shared__ float    aL[EPG * HEADS];
  const int b = blockIdx.x, tid = threadIdx.x;
  for (int i = tid; i < NPG * HEADS; i += blockDim.x) { mx[i] = 0u; sm[i] = 0.f; }
  __syncthreads();
  const int e0 = b * EPG;
  // pass 1: segment max (order-preserving uint atomics)
  for (int p = tid; p < EPG * HEADS; p += blockDim.x) {
    const int el = p >> 3, hh = p & 7;
    const int dl = dst[e0 + el] - b * NPG;
    atomicMax(&mx[dl * HEADS + hh], f2ou(logits[(e0 + el) * HEADS + hh]));
  }
  __syncthreads();
  // pass 2: exp + segment sum
  for (int p = tid; p < EPG * HEADS; p += blockDim.x) {
    const int el = p >> 3, hh = p & 7;
    const int dl = dst[e0 + el] - b * NPG;
    const float ex = __expf(logits[(e0 + el) * HEADS + hh] - ou2f(mx[dl * HEADS + hh]));
    aL[p] = ex;
    atomicAdd(&sm[dl * HEADS + hh], ex);
  }
  __syncthreads();
  // pass 3: normalize + write out
  for (int p = tid; p < EPG * HEADS; p += blockDim.x) {
    const int el = p >> 3, hh = p & 7;
    const int dl = dst[e0 + el] - b * NPG;
    const float v = aL[p] / sm[dl * HEADS + hh];
    aL[p] = v;
    att[(e0 + el) * HEADS + hh] = v;
  }
  __syncthreads();
  // virtual edges are the last NATOM edges of each molecule (== attn_eids)
  if (tid < NATOM) {
    const int el = EINTRA + tid;
    float s = 0.f;
#pragma unroll
    for (int hh = 0; hh < HEADS; ++hh) s += aL[el * HEADS + hh];
    attn_out_t[b * NATOM + tid] = s * (1.0f / HEADS);
  }
}

// ---------------------------------------------------------------- aggregation
// hnew[n,d] = (1/8) * sum_h sum_{e: dst=n} att[e,h] * fs[src(e),h,d]
// grid (BATCH, 3) : one molecule x one 256-dim chunk per block
__global__ void aggregate_kernel(const float* __restrict__ att,
                                 const float* __restrict__ fs,
                                 const int* __restrict__ src,
                                 const int* __restrict__ dst,
                                 float* __restrict__ hnew) {
  __shared__ float acc[NPG][256];   // each thread owns one dim column -> no atomics
  const int b = blockIdx.x, dbase = blockIdx.y * 256, tid = threadIdx.x;
  for (int n = 0; n < NPG; ++n) acc[n][tid] = 0.f;
  const int e0 = b * EPG;
  for (int el = 0; el < EPG; ++el) {
    const int e  = e0 + el;
    const int sg = src[e];
    const int dl = dst[e] - b * NPG;
    float v = 0.f;
#pragma unroll
    for (int hh = 0; hh < HEADS; ++hh)
      v += att[e * HEADS + hh] * fs[((long)sg * HEADS + hh) * H + dbase + tid];
    acc[dl][tid] += v;
  }
  const int nbase = b * NPG;
  for (int n = 0; n < NPG; ++n)
    hnew[(long)(nbase + n) * H + dbase + tid] = acc[n][tid] * (1.0f / HEADS);
}

// ---------------------------------------------------------------- GRU gate GEMVs
// grid (9, BATCH, 2): z=0 -> gi = x @ W_ih^T + b_ih (x = hnew[virtual]);
//                     z=1 -> gh = hprev @ W_hh^T + b_hh (hprev = molemb)
__global__ void gru_gemv_kernel(const float* __restrict__ hnew,
                                const int* __restrict__ vnids,
                                const float* __restrict__ molemb,
                                const float* __restrict__ W_ih_t,
                                const float* __restrict__ W_hh_t,
                                const float* __restrict__ b_ih_t,
                                const float* __restrict__ b_hh_t,
                                float* __restrict__ gi,
                                float* __restrict__ gh) {
  __shared__ float xs[H];
  const int b = blockIdx.y;
  const int o = blockIdx.x * 256 + threadIdx.x;          // 0..2303
  const float* x;  const float* W; const float* bias; float* out;
  if (blockIdx.z == 0) { x = hnew + (long)vnids[b] * H; W = W_ih_t; bias = b_ih_t; out = gi; }
  else                 { x = molemb + (long)b * H;      W = W_hh_t; bias = b_hh_t; out = gh; }
  for (int k = threadIdx.x; k < H; k += 256) xs[k] = x[k];
  __syncthreads();
  float a = bias[o];
  const float* wr = W + (long)o * H;
  for (int k = 0; k < H; ++k) a += xs[k] * wr[k];
  out[(long)b * G3H + o] = a;
}

// ---------------------------------------------------------------- GRU update + relu
__global__ void gru_update_kernel(const float* __restrict__ gi,
                                  const float* __restrict__ gh,
                                  float* __restrict__ molemb,
                                  float* __restrict__ out_final) {
  const int idx = blockIdx.x * 256 + threadIdx.x;
  if (idx >= BATCH * H) return;
  const int b = idx / H, d = idx % H;
  const float* gib = gi + (long)b * G3H;
  const float* ghb = gh + (long)b * G3H;
  const float r  = sigmoidf(gib[d]         + ghb[d]);
  const float z  = sigmoidf(gib[H + d]     + ghb[H + d]);
  const float n  = tanhf   (gib[2 * H + d] + r * ghb[2 * H + d]);
  const float hp = molemb[idx];
  float hn = (1.0f - z) * n + z * hp;
  hn = hn > 0.f ? hn : 0.f;                 // relu
  molemb[idx] = hn;
  if (out_final) out_final[idx] = hn;
}

// ---------------------------------------------------------------- launch
extern "C" void kernel_launch(void* const* d_in, const int* in_sizes, int n_in,
                              void* d_out, int out_size, void* d_ws, size_t ws_size,
                              hipStream_t stream) {
  const float* h_nodes  = (const float*)d_in[0];
  const float* mol_feat = (const float*)d_in[1];
  const float* W_src    = (const float*)d_in[2];
  const float* b_src    = (const float*)d_in[3];
  const float* W_dst    = (const float*)d_in[4];
  const float* b_dst    = (const float*)d_in[5];
  const float* attn_a   = (const float*)d_in[6];
  const float* W_ih     = (const float*)d_in[7];
  const float* W_hh     = (const float*)d_in[8];
  const float* b_ih     = (const float*)d_in[9];
  const float* b_hh     = (const float*)d_in[10];
  const int*   src      = (const int*)d_in[11];
  const int*   dst      = (const int*)d_in[12];
  const int*   vnids    = (const int*)d_in[13];
  float* out = (float*)d_out;   // [BATCH*H mol_emb][T * BATCH*NATOM attns]

  // ---- workspace carve (256B aligned)
  char* p = (char*)d_ws;
  auto carve = [&p](size_t bytes) -> void* {
    void* r = (void*)p; p += (bytes + 255) & ~(size_t)255; return r;
  };
  const long wsz = (long)T * H * MOUT;          // weight elements per tensor
  unsigned short* Wsrc16 = (unsigned short*)carve(wsz * 2);   // packed fragments
  unsigned short* Wdst16 = (unsigned short*)carve(wsz * 2);   // packed fragments
  unsigned short* h16    = (unsigned short*)carve((long)NTOT * H * 2);
  float* fs     = (float*)carve((long)NTOT * MOUT * 4);
  float* fd     = (float*)carve((long)NTOT * MOUT * 4);
  float* logits = (float*)carve((long)ETOT * HEADS * 4);
  float* att    = (float*)carve((long)ETOT * HEADS * 4);
  float* hnew   = (float*)carve((long)NTOT * H * 4);
  float* gi     = (float*)carve((long)BATCH * G3H * 4);
  float* gh     = (float*)carve((long)BATCH * G3H * 4);
  float* molemb = (float*)carve((long)BATCH * H * 4);

  // ---- one-time: weights -> bf16 packed B-fragment tiles ; molemb <- mol_feat
  pack_B_kernel<<<(int)((wsz + 255) / 256), 256, 0, stream>>>(W_src, Wsrc16, wsz);
  pack_B_kernel<<<(int)((wsz + 255) / 256), 256, 0, stream>>>(W_dst, Wdst16, wsz);
  hipMemcpyAsync(molemb, mol_feat, (size_t)BATCH * H * sizeof(float),
                 hipMemcpyDeviceToDevice, stream);

  const float* hcur = h_nodes;
  for (int t = 0; t < T; ++t) {
    const long nh = (long)NTOT * H;
    cvt_bf16_kernel<<<(int)((nh + 255) / 256), 256, 0, stream>>>(hcur, h16, nh);

    dim3 gg(MOUT / 128, NTOT / 128, 2);   // (48, 49, 2)
    gemm_bf16_kernel<<<gg, 256, 0, stream>>>(
        h16, Wsrc16 + (long)t * H * MOUT, Wdst16 + (long)t * H * MOUT,
        b_src + (long)t * MOUT, b_dst + (long)t * MOUT, fs, fd);

    edge_logits_kernel<<<(ETOT * HEADS) / 8, 256, 0, stream>>>(
        fs, fd, attn_a + (long)t * HEADS * H, src, dst, logits);

    softmax_kernel<<<BATCH, 256, 0, stream>>>(
        logits, dst, att, out + BATCH * H + (long)t * BATCH * NATOM);

    aggregate_kernel<<<dim3(BATCH, H / 256), 256, 0, stream>>>(att, fs, src, dst, hnew);

    gru_gemv_kernel<<<dim3(G3H / 256, BATCH, 2), 256, 0, stream>>>(
        hnew, vnids, molemb,
        W_ih + (long)t * G3H * H, W_hh + (long)t * G3H * H,
        b_ih + (long)t * G3H, b_hh + (long)t * G3H, gi, gh);

    gru_update_kernel<<<(BATCH * H + 255) / 256, 256, 0, stream>>>(
        gi, gh, molemb, (t == T - 1) ? out : nullptr);

    hcur = hnew;
  }
}
